// SINQLinear_26645977104586
// MI455X (gfx1250) — matrix operand order repaired and checked by reading, not compiled, not run on hardware
//
#include <hip/hip_runtime.h>

// SINQ int4 linear for MI455X (gfx1250): f16 WMMA + TDM (tensor_load_to_lds)
// out[m,n] = sum_k (x[m,k]*scale2[k]) * ((code[n,k]-zero[n,g])*scale[n,g]) + bias[n]

typedef __attribute__((ext_vector_type(16))) _Float16 v16h;
typedef __attribute__((ext_vector_type(8)))  _Float16 v8h;
typedef __attribute__((ext_vector_type(8)))  float    v8f;
typedef unsigned int u32x4 __attribute__((ext_vector_type(4)));
typedef int          i32x8 __attribute__((ext_vector_type(8)));
typedef int          i32x4 __attribute__((ext_vector_type(4)));

#define M_DIM 8192
#define N_DIM 11008
#define K_DIM 4096
#define G_DIM 128
#define KG    (K_DIM / G_DIM)    // 32 groups
#define KSTEP 32
#define NSTEPS (K_DIM / KSTEP)   // 128
#define TSTRIDE 40               // halves: 32 data + 8 pad = 80 B row, conflict-free b128

union V16U { v16h v; v8h p[2]; };

// ---------------------------------------------------------------------------
// Phase 1: xs[m,k] = (half)(x[m,k] * scale2[k]).  8 elements per thread.
// ---------------------------------------------------------------------------
__global__ __launch_bounds__(256)
void sinq_prep_kernel(const float* __restrict__ x,
                      const float* __restrict__ scale2,
                      _Float16* __restrict__ xs)
{
    const long i   = ((long)blockIdx.x * 256 + threadIdx.x) * 8;
    const int  col = (int)(i & (K_DIM - 1));

    const float4 a  = *(const float4*)(x + i);
    const float4 b  = *(const float4*)(x + i + 4);
    const float4 s0 = *(const float4*)(scale2 + col);
    const float4 s1 = *(const float4*)(scale2 + col + 4);

    v8h r;
    r[0] = (_Float16)(a.x * s0.x);
    r[1] = (_Float16)(a.y * s0.y);
    r[2] = (_Float16)(a.z * s0.z);
    r[3] = (_Float16)(a.w * s0.w);
    r[4] = (_Float16)(b.x * s1.x);
    r[5] = (_Float16)(b.y * s1.y);
    r[6] = (_Float16)(b.z * s1.z);
    r[7] = (_Float16)(b.w * s1.w);
    *(v8h*)(xs + i) = r;
}

// ---------------------------------------------------------------------------
// TDM: DMA one 128-row x 32-half A-tile (row stride K) from global into LDS,
// padding each 64 B row with 16 B (pad_interval=16 DWORDs, pad_amount=4 DWORDs)
// so the LDS row stride is 80 B (bank-conflict-free for b128 fragment reads).
// D# built per CDNA5 ISA §8 (group0: count/lds/global/type, group1: dims).
// This toolchain declares the 6-arg builtin:
//   (u32x4 g0, i32x8 g1, i32x4 g2, i32x4 g3, i32x8 extra, i32 cpol)
// ---------------------------------------------------------------------------
__device__ __forceinline__ void tdm_load_a_tile(const _Float16* gsrc,
                                                const _Float16* ldst)
{
    const unsigned long long ga  = (unsigned long long)(size_t)gsrc;
    const unsigned           lds = (unsigned)(size_t)ldst;

    u32x4 g0;
    g0[0] = 1u;                                   // count=1 (valid descriptor)
    g0[1] = lds;                                  // lds_addr (bytes)
    g0[2] = (unsigned)ga;                         // global_addr[31:0]
    g0[3] = (unsigned)((ga >> 32) & 0x1FFFFFFu)   // global_addr[56:32]
          | (2u << 30);                           // type=2 ("image")

    i32x8 g1;
    g1[0] = (1 << 16)        // data_size = 1 -> 2 bytes
          | (1 << 20)        // pad_enable
          | (3 << 22)        // pad_interval: 2^(3+1)=16 DWORDs = 64 B
          | (3 << 25);       // pad_amount: 3+1 = 4 DWORDs = 16 B
    g1[1] = (K_DIM & 0xFFFF) << 16;   // tensor_dim0 = 4096 (low 16)
    g1[2] = (M_DIM & 0xFFFF) << 16;   // tensor_dim1 = 8192 (low 16)
    g1[3] = KSTEP << 16;              // tile_dim0 = 32 elements
    g1[4] = 128;                      // tile_dim1 = 128 rows (tile_dim2 = 0)
    g1[5] = K_DIM;                    // tensor_dim0_stride = 4096 elements
    g1[6] = 0;
    g1[7] = 0;

    const i32x4 gz4 = {0, 0, 0, 0};             // groups 2/3 unused (2-D tensor)
    const i32x8 gz8 = {0, 0, 0, 0, 0, 0, 0, 0};
    __builtin_amdgcn_tensor_load_to_lds(g0, g1, gz4, gz4, gz8, 0);
}

// ---------------------------------------------------------------------------
// int4 -> f16 dequant of one 128x32 W tile into LDS (16 codes per thread).
// ---------------------------------------------------------------------------
__device__ __forceinline__ void dequant_w_tile(const int* __restrict__ wq,
                                               long wqRow, int nRow, int s,
                                               const float* __restrict__ scale,
                                               const float* __restrict__ zero,
                                               int dr, int dh,
                                               _Float16* __restrict__ wbuf)
{
    const int   k0 = s * KSTEP;
    const int   g  = k0 >> 7;                       // group of 128
    const float sc = scale[(long)nRow * KG + g];
    const float zp = zero [(long)nRow * KG + g];
    const int*  wp = wq + wqRow + (k0 >> 1) + dh * 8;

    _Float16 tmp[16];
    #pragma unroll
    for (int i = 0; i < 8; ++i) {
        const int b = wp[i];
        tmp[2 * i]     = (_Float16)(((float)(b & 0xF)        - zp) * sc);
        tmp[2 * i + 1] = (_Float16)(((float)((b >> 4) & 0xF) - zp) * sc);
    }
    _Float16* dst = wbuf + dr * TSTRIDE + dh * 16;
    *(v8h*)dst       = *(const v8h*)&tmp[0];
    *(v8h*)(dst + 8) = *(const v8h*)&tmp[8];
}

// ---------------------------------------------------------------------------
// Phase 2: double-buffered WMMA GEMM.
//   Block: 256 threads = 8 wave32s.  Block tile 128x128, wave tile 32x64.
//   A tile: TDM async DMA (TENSORcnt).  W tile: cooperative int4 dequant.
// ---------------------------------------------------------------------------
__global__ __launch_bounds__(256)
void sinq_wmma_gemm_kernel(const _Float16* __restrict__ xs,
                           const int*      __restrict__ wq,     // [N, K/2] byte-per-int
                           const float*    __restrict__ scale,  // [N, K/G]
                           const float*    __restrict__ zero,   // [N, K/G]
                           const float*    __restrict__ bias,   // [N]
                           float*          __restrict__ out)    // [M, N]
{
    __shared__ _Float16 As[2][128 * TSTRIDE];   // 2 x 10 KB
    __shared__ _Float16 Ws[2][128 * TSTRIDE];   // 2 x 10 KB

    const int tid  = threadIdx.x;
    const int lane = tid & 31;
    const int wave = tid >> 5;
    const int l    = lane & 15;
    const int h    = lane >> 4;
    const int wm   = wave & 3;      // 4 M sub-tiles of 32
    const int wn   = wave >> 2;     // 2 N sub-tiles of 64

    const int mBlock = blockIdx.y * 128;
    const int nBlock = blockIdx.x * 128;

    const int  dr    = tid >> 1;    // dequant: local W row
    const int  dh    = tid & 1;     // dequant: 16-code half
    const int  nRow  = nBlock + dr;
    const long wqRow = (long)nRow * (K_DIM / 2);

    const _Float16* aGlobalBase = xs + (long)mBlock * K_DIM;

    v8f acc[2][4] = {};

    // ---- prologue: stage tile 0 into buffer 0 ----------------------------
    if (wave == 0)
        tdm_load_a_tile(aGlobalBase, &As[0][0]);
    dequant_w_tile(wq, wqRow, nRow, 0, scale, zero, dr, dh, &Ws[0][0]);
    __builtin_amdgcn_s_wait_tensorcnt(0);
    __syncthreads();

    for (int s = 0; s < NSTEPS; ++s) {
        const int b  = s & 1;
        const int nb = b ^ 1;

        // ---- stage tile s+1 into the alternate buffers -------------------
        if (s + 1 < NSTEPS) {
            if (wave == 0)
                tdm_load_a_tile(aGlobalBase + (long)(s + 1) * KSTEP, &As[nb][0]);
            dequant_w_tile(wq, wqRow, nRow, s + 1, scale, zero, dr, dh, &Ws[nb][0]);
        }

        // ---- A fragments from LDS (TDM-padded rows, stride 40 halves) ----
        // A k order used here (same as B): lanes 0-15 K=0..15,
        //                        lanes 16-31 K=16..31.
        v16h afrag[2];
        #pragma unroll
        for (int mi = 0; mi < 2; ++mi) {
            const _Float16* ap = &As[b][(wm * 32 + mi * 16 + l) * TSTRIDE + h * 16];
            V16U u;
            u.p[0] = *(const v8h*)ap;
            u.p[1] = *(const v8h*)(ap + 8);
            afrag[mi] = u.v;
        }

        // ---- B fragments from LDS ---------------------------------------
        // B k order used here (same as A): lanes 0-15 K=0..15, lanes 16-31 K=16..31.
        v16h bfrag[4];
        #pragma unroll
        for (int ni = 0; ni < 4; ++ni) {
            const _Float16* bp = &Ws[b][(wn * 64 + ni * 16 + l) * TSTRIDE + h * 16];
            V16U u;
            u.p[0] = *(const v8h*)bp;
            u.p[1] = *(const v8h*)(bp + 8);
            bfrag[ni] = u.v;
        }

        // ---- matrix core: 8 x v_wmma_f32_16x16x32_f16 --------------------
        #pragma unroll
        for (int mi = 0; mi < 2; ++mi)
            #pragma unroll
            for (int ni = 0; ni < 4; ++ni)
                acc[mi][ni] = __builtin_amdgcn_wmma_f32_16x16x32_f16(
                    false, afrag[mi], false, bfrag[ni],
                    (short)0, acc[mi][ni], false, false);

        // ---- publish next buffers (TDM done + dequant stores visible) ----
        __builtin_amdgcn_s_wait_tensorcnt(0);
        __syncthreads();
    }

    // ---- epilogue: bias + store per the 16x16 f32 C/D layout --------------
    // VGPR i: lanes 0-15 -> (M=i, N=l); lanes 16-31 -> (M=i+8, N=l).
    #pragma unroll
    for (int ni = 0; ni < 4; ++ni) {
        const int   n  = nBlock + wn * 64 + ni * 16 + l;
        const float bv = bias[n];
        #pragma unroll
        for (int mi = 0; mi < 2; ++mi) {
            #pragma unroll
            for (int i = 0; i < 8; ++i) {
                const int m = mBlock + wm * 32 + mi * 16 + h * 8 + i;
                out[(long)m * N_DIM + n] = acc[mi][ni][i] + bv;
            }
        }
    }
}

// ---------------------------------------------------------------------------
extern "C" void kernel_launch(void* const* d_in, const int* in_sizes, int n_in,
                              void* d_out, int out_size, void* d_ws, size_t ws_size,
                              hipStream_t stream)
{
    const float* x      = (const float*)d_in[0];
    const int*   wq     = (const int*)  d_in[1];
    const float* scale  = (const float*)d_in[2];
    const float* zero   = (const float*)d_in[3];
    const float* scale2 = (const float*)d_in[4];
    const float* bias   = (const float*)d_in[5];
    float*       out    = (float*)d_out;

    _Float16* xs = (_Float16*)d_ws;   // 8192*4096 halves = 64 MB scratch

    {
        const long total  = (long)M_DIM * K_DIM;
        const int  blocks = (int)(total / (256 * 8));   // 16384
        sinq_prep_kernel<<<blocks, 256, 0, stream>>>(x, scale2, xs);
    }
    {
        dim3 grid(N_DIM / 128, M_DIM / 128);            // 86 x 64
        sinq_wmma_gemm_kernel<<<grid, 256, 0, stream>>>(xs, wq, scale, zero, bias, out);
    }
}
